// BasicLayer_40321152975591
// MI455X (gfx1250) — matrix operand order repaired and checked by reading, not compiled, not run on hardware
//
#include <hip/hip_runtime.h>
#include <hip/hip_bf16.h>

typedef __attribute__((ext_vector_type(16))) __bf16 v16bf;
typedef __attribute__((ext_vector_type(8)))  float v8f;

struct B32x2 { uint4 lo, hi; };

__device__ __forceinline__ unsigned short f2bf(float f) {
    unsigned int u = __float_as_uint(f);
    u += 0x7FFFu + ((u >> 16) & 1u);   // round-to-nearest-even
    return (unsigned short)(u >> 16);
}

__device__ __forceinline__ unsigned int pack2_bf(float a, float b) {
    return (unsigned int)f2bf(a) | ((unsigned int)f2bf(b) << 16);
}

__device__ __forceinline__ v8f v8f_zero() {
    v8f z = {0.f, 0.f, 0.f, 0.f, 0.f, 0.f, 0.f, 0.f};
    return z;
}

// A operand (16x32 bf16): lane L holds row rowBase+(L&15);
// lanes 0-15: K = {0..7, 16..23}, lanes 16-31: K = {8..15, 24..31} (rel. kBase)
__device__ __forceinline__ v16bf load_matA(const unsigned short* buf, int stride,
                                           int rowBase, int kBase, int lane) {
    int row = rowBase + (lane & 15);
    int k0  = kBase + ((lane >> 4) << 3);
    const unsigned short* p = buf + row * stride + k0;
    B32x2 d;
    d.lo = *(const uint4*)(p);
    d.hi = *(const uint4*)(p + 16);
    return __builtin_bit_cast(v16bf, d);
}

// B operand (32x16 bf16) where each logical B-column is a contiguous row
// (works for LDS and for pre-converted bf16 global weights):
// lane L holds column rowBase+(L&15); lanes 0-15: K = {0..7, 16..23}, lanes 16-31: K = {8..15, 24..31}
__device__ __forceinline__ v16bf load_matB(const unsigned short* buf, int stride,
                                           int rowBase, int kBase, int lane) {
    int row = rowBase + (lane & 15);
    int k0  = kBase + ((lane >> 4) << 3);
    const unsigned short* p = buf + row * stride + k0;
    B32x2 d;
    d.lo = *(const uint4*)(p);
    d.hi = *(const uint4*)(p + 16);
    return __builtin_bit_cast(v16bf, d);
}

__device__ __forceinline__ v8f wmma_bf16(v16bf a, v16bf b, v8f c) {
    return __builtin_amdgcn_wmma_f32_16x16x32_bf16(false, a, false, b, (short)0, c,
                                                   false, false);
}

__device__ __forceinline__ int regcode(int y) {
    // shift=3, ws=7, H=112: regions [0,105) [105,109) [109,112)
    return (y < 105) ? 0 : ((y < 109) ? 1 : 2);
}

// ---------------------------------------------------------------------------
// One-shot weight conversion f32 -> bf16 (q rows pre-scaled by 1/sqrt(hd)).
// Weights are identical for all 2048 windows; convert once, not per block.
// ---------------------------------------------------------------------------
#define NQKV (2 * 1152 * 384)
#define NPROJ (2 * 384 * 384)

__global__ __launch_bounds__(256) void convert_weights(
    const float* __restrict__ qkvW, const float* __restrict__ projW,
    unsigned short* __restrict__ qkvWbf, unsigned short* __restrict__ projWbf) {
    int i = blockIdx.x * 256 + threadIdx.x;
    if (i < NQKV) {
        int rem = i % (1152 * 384);
        float v = qkvW[i];
        if (rem < 384 * 384) v *= 0.17677669529663689f;  // q rows: fold 1/sqrt(32)
        qkvWbf[i] = f2bf(v);
    }
    if (i < NPROJ) projWbf[i] = f2bf(projW[i]);
}

// LDS layout (halfword units):
//   xbuf : 64 rows x 392   (x window bf16, reused as attention output)
//   qkbuf: 64 rows x 776   (q cols 0..383 prescaled, k cols 384..767)
//   vtbuf: 12 heads x 32 x 72  (v transposed: [head][hd_elem][key])
//   pbuf : 8 waves x 64 x 72   (softmax probabilities, per-wave)
#define XS   392
#define QKS  776
#define VTS  72
#define PS   72
#define SMEM_HALFWORDS (64 * XS + 64 * QKS + 12 * 32 * VTS + 8 * 64 * PS)

template <int SHIFT, bool INBF, bool OUTBF>
__global__ __launch_bounds__(256, 1) void swin_layer(
    const void* __restrict__ x_in_v, const unsigned short* __restrict__ qkvW,
    const float* __restrict__ qkvB, const unsigned short* __restrict__ projW,
    const float* __restrict__ projB, const float* __restrict__ rpb,
    void* __restrict__ x_out_v) {
    extern __shared__ unsigned short smem[];
    unsigned short* xbuf  = smem;
    unsigned short* qkbuf = smem + 64 * XS;
    unsigned short* vtbuf = qkbuf + 64 * QKS;
    unsigned short* pbuf  = vtbuf + 12 * 32 * VTS;

    const int tid  = threadIdx.x;
    const int lane = tid & 31;
    const int wv   = tid >> 5;
    const int win  = blockIdx.x;
    const int b    = win >> 8;       // 256 windows per image
    const int wi   = win & 255;
    const int wh   = wi >> 4, ww = wi & 15;
    const int half8 = (lane >> 4) << 3;  // row offset for C/D lanes 16-31

    // ---- Phase 0: gather x window (with roll) -> LDS bf16, zero pad rows ----
    for (int i = tid; i < 49 * 96; i += 256) {
        int row = i / 96, c4 = (i - row * 96) * 4;
        int ty = row / 7, tx = row - ty * 7;
        int ih = wh * 7 + ty + SHIFT; if (ih >= 112) ih -= 112;
        int iw = ww * 7 + tx + SHIFT; if (iw >= 112) iw -= 112;
        size_t tok = (size_t)(b * 12544 + ih * 112 + iw) * 384 + c4;
        unsigned int* dst = (unsigned int*)(xbuf + row * XS + c4);
        if (INBF) {
            const uint2 u = *(const uint2*)((const unsigned short*)x_in_v + tok);
            dst[0] = u.x;
            dst[1] = u.y;
        } else {
            const float4 f = *(const float4*)((const float*)x_in_v + tok);
            dst[0] = pack2_bf(f.x, f.y);
            dst[1] = pack2_bf(f.z, f.w);
        }
    }
    for (int i = tid; i < 15 * XS; i += 256) {
        int row = 49 + i / XS;
        xbuf[row * XS + (i % XS)] = 0;
    }
    __syncthreads();

    // ---- Phase 1: QKV GEMM  [64x384] x [384x1152]  (bf16 WMMA, f32 acc) ----
    const float qscale = 0.17677669529663689f;  // 1/sqrt(32) (weights prescaled)
    for (int nt = wv; nt < 72; nt += 8) {
        int o0 = nt * 16;
        int o  = o0 + (lane & 15);
        v8f acc[4] = {v8f_zero(), v8f_zero(), v8f_zero(), v8f_zero()};
        for (int ks = 0; ks < 12; ++ks) {
            if (ks < 11)  // cover L2 latency on the streamed bf16 weights
                __builtin_prefetch(qkvW + (size_t)o * 384 + (ks + 1) * 32, 0, 1);
            v16bf bm = load_matB(qkvW, 384, o0, ks * 32, lane);
#pragma unroll
            for (int mt = 0; mt < 4; ++mt) {
                v16bf am = load_matA(xbuf, XS, mt * 16, ks * 32, lane);
                acc[mt] = wmma_bf16(am, bm, acc[mt]);
            }
        }
        float bias = qkvB[o];
        if (o0 < 384) bias *= qscale;  // q bias picks up the folded scale
        if (o0 < 768) {                // tile-uniform: whole tile is q or k
#pragma unroll
            for (int mt = 0; mt < 4; ++mt) {
#pragma unroll
                for (int r = 0; r < 8; ++r) {
                    int m = mt * 16 + r + half8;
                    qkbuf[m * QKS + o] = f2bf(acc[mt][r] + bias);
                }
            }
        } else {  // tile-uniform: whole tile is v -> store transposed
            int e = o - 768;
            unsigned short* vrow = vtbuf + (e >> 5) * (32 * VTS) + (e & 31) * VTS;
#pragma unroll
            for (int mt = 0; mt < 4; ++mt) {
#pragma unroll
                for (int r = 0; r < 8; ++r) {
                    int m = mt * 16 + r + half8;
                    vrow[m] = f2bf(acc[mt][r] + bias);
                }
            }
        }
    }
    __syncthreads();

    // ---- Phase 2: attention per head ----
    for (int h = wv; h < 12; h += 8) {
        unsigned short* pb = pbuf + wv * (64 * PS);
        const float* rpbh = rpb + h;
        for (int mt = 0; mt < 4; ++mt) {
            v8f s[4] = {v8f_zero(), v8f_zero(), v8f_zero(), v8f_zero()};
            v16bf aq = load_matA(qkbuf, QKS, mt * 16, h * 32, lane);
#pragma unroll
            for (int nt = 0; nt < 4; ++nt) {
                v16bf bk = load_matB(qkbuf, QKS, nt * 16, 384 + h * 32, lane);
                s[nt] = wmma_bf16(aq, bk, s[nt]);
            }
            int mrowbase = mt * 16 + half8;
            int n0 = lane & 15;
#pragma unroll
            for (int r = 0; r < 8; ++r) {
                int m = mrowbase + r;
                bool mval = m < 49;
                int my = m / 7, mx = m - my * 7;
                float sv[4];
#pragma unroll
                for (int nt = 0; nt < 4; ++nt) {
                    int n = nt * 16 + n0;
                    if (mval && n < 49) {
                        int ny = n / 7, nx = n - ny * 7;
                        float bias = rpbh[(((my - ny + 6) * 13) + (mx - nx + 6)) * 12];
                        float mterm = 0.f;
                        if (SHIFT > 0) {
                            int rm = regcode(wh * 7 + my) * 3 + regcode(ww * 7 + mx);
                            int rn = regcode(wh * 7 + ny) * 3 + regcode(ww * 7 + nx);
                            mterm = (rm != rn) ? -100.f : 0.f;
                        }
                        sv[nt] = s[nt][r] + bias + mterm;
                    } else {
                        sv[nt] = -1e30f;
                    }
                }
                float rmax = fmaxf(fmaxf(sv[0], sv[1]), fmaxf(sv[2], sv[3]));
#pragma unroll
                for (int off = 1; off < 16; off <<= 1)
                    rmax = fmaxf(rmax, __shfl_xor(rmax, off, 32));
                float ev[4];
                float rsum = 0.f;
#pragma unroll
                for (int nt = 0; nt < 4; ++nt) {
                    ev[nt] = __expf(sv[nt] - rmax);
                    rsum += ev[nt];
                }
#pragma unroll
                for (int off = 1; off < 16; off <<= 1)
                    rsum += __shfl_xor(rsum, off, 32);
                float rinv = 1.f / rsum;
#pragma unroll
                for (int nt = 0; nt < 4; ++nt) {
                    int n = nt * 16 + n0;
                    float p = (mval && n < 49) ? ev[nt] * rinv : 0.f;
                    pb[m * PS + n] = f2bf(p);
                }
            }
        }
        // O = P @ V  (K=64 keys, N=32 hd)
        for (int mt = 0; mt < 4; ++mt) {
#pragma unroll
            for (int ntv = 0; ntv < 2; ++ntv) {
                v8f oacc = v8f_zero();
#pragma unroll
                for (int kt = 0; kt < 2; ++kt) {
                    v16bf ap = load_matA(pb, PS, mt * 16, kt * 32, lane);
                    v16bf bv = load_matB(vtbuf + h * (32 * VTS), VTS, ntv * 16,
                                         kt * 32, lane);
                    oacc = wmma_bf16(ap, bv, oacc);
                }
#pragma unroll
                for (int r = 0; r < 8; ++r) {
                    int m = mt * 16 + r + half8;
                    if (m < 49)
                        xbuf[m * XS + h * 32 + ntv * 16 + (lane & 15)] = f2bf(oacc[r]);
                }
            }
        }
    }
    __syncthreads();

    // ---- Phase 3: proj GEMM [64x384] x [384x384], scatter with window-reverse ----
    for (int nt = wv; nt < 24; nt += 8) {
        int o0 = nt * 16;
        int o  = o0 + (lane & 15);
        v8f acc[4] = {v8f_zero(), v8f_zero(), v8f_zero(), v8f_zero()};
        for (int ks = 0; ks < 12; ++ks) {
            if (ks < 11)
                __builtin_prefetch(projW + (size_t)o * 384 + (ks + 1) * 32, 0, 1);
            v16bf bm = load_matB(projW, 384, o0, ks * 32, lane);
#pragma unroll
            for (int mt = 0; mt < 4; ++mt) {
                v16bf am = load_matA(xbuf, XS, mt * 16, ks * 32, lane);
                acc[mt] = wmma_bf16(am, bm, acc[mt]);
            }
        }
        float pbias = projB[o];
#pragma unroll
        for (int mt = 0; mt < 4; ++mt) {
#pragma unroll
            for (int r = 0; r < 8; ++r) {
                int m = mt * 16 + r + half8;
                if (m < 49) {
                    int ty = m / 7, tx = m - ty * 7;
                    int ih = wh * 7 + ty + SHIFT; if (ih >= 112) ih -= 112;
                    int iw = ww * 7 + tx + SHIFT; if (iw >= 112) iw -= 112;
                    size_t idx = (size_t)(b * 12544 + ih * 112 + iw) * 384 + o;
                    float val = acc[mt][r] + pbias;
                    if (OUTBF)
                        ((unsigned short*)x_out_v)[idx] = f2bf(val);
                    else
                        ((float*)x_out_v)[idx] = val;
                }
            }
        }
    }
}

extern "C" void kernel_launch(void* const* d_in, const int* in_sizes, int n_in,
                              void* d_out, int out_size, void* d_ws, size_t ws_size,
                              hipStream_t stream) {
    (void)in_sizes; (void)n_in; (void)out_size; (void)ws_size;
    const float* x      = (const float*)d_in[0];
    const float* qkv_w  = (const float*)d_in[1];  // [2, 1152, 384]
    const float* qkv_b  = (const float*)d_in[2];  // [2, 1152]
    const float* proj_w = (const float*)d_in[3];  // [2, 384, 384]
    const float* proj_b = (const float*)d_in[4];  // [2, 384]
    const float* rpb    = (const float*)d_in[5];  // [2, 169, 12]

    // workspace layout
    unsigned short* x1bf    = (unsigned short*)d_ws;            // 8*12544*384 bf16
    unsigned short* qkvWbf  = x1bf + (size_t)8 * 12544 * 384;   // [2,1152,384] bf16
    unsigned short* projWbf = qkvWbf + NQKV;                    // [2,384,384]  bf16

    const size_t shmem = (size_t)SMEM_HALFWORDS * sizeof(unsigned short);
    (void)hipFuncSetAttribute((const void*)swin_layer<0, false, true>,
                              hipFuncAttributeMaxDynamicSharedMemorySize, (int)shmem);
    (void)hipFuncSetAttribute((const void*)swin_layer<3, true, false>,
                              hipFuncAttributeMaxDynamicSharedMemorySize, (int)shmem);

    // one-shot weight conversion (q rows prescaled by 1/sqrt(hd))
    convert_weights<<<dim3((NQKV + 255) / 256), dim3(256), 0, stream>>>(
        qkv_w, proj_w, qkvWbf, projWbf);

    dim3 grid(2048), block(256);
    // depth 0: shift = 0, f32 in, bf16 out (inter-layer buffer)
    swin_layer<0, false, true><<<grid, block, shmem, stream>>>(
        x, qkvWbf, qkv_b, projWbf, proj_b, rpb, x1bf);
    // depth 1: shift = 3, bf16 in, f32 out
    swin_layer<3, true, false><<<grid, block, shmem, stream>>>(
        x1bf, qkvWbf + 1152 * 384, qkv_b + 1152, projWbf + 384 * 384,
        proj_b + 384, rpb + 169 * 12, d_out);
}
